// VmfQuantizer_38697655337583
// MI455X (gfx1250) — matrix operand order repaired and checked by reading, not compiled, not run on hardware
//
#include <hip/hip_runtime.h>
#include <hip/hip_fp16.h>

typedef __attribute__((ext_vector_type(16))) _Float16 v16h;
typedef __attribute__((ext_vector_type(8)))  float    v8f;

#define BATCH   32
#define DCH     64
#define HH      64
#define WW      64
#define KCB     512
#define NTOK    (BATCH*HH*WW)        // 131072 tokens
#define HWSZ    (HH*WW)              // 4096
#define ZQ_ELEMS (BATCH*DCH*HH*WW)   // 8388608
#define REG_OFF  ZQ_ELEMS
#define IDX_OFF  (ZQ_ELEMS + 1)
#define CHUNK_CODES 256              // codes staged in LDS per chunk (32KB)

// ---------------------------------------------------------------------------
// counter-based hash -> uniform(0,1); deterministic per counter value
// (fixed-shift finalizer: cheaper than PCG's variable shift on VALU)
// ---------------------------------------------------------------------------
__device__ __forceinline__ float u01_hash(unsigned s) {
    s ^= s >> 16;  s *= 0x7FEB352Du;
    s ^= s >> 15;  s *= 0x846CA68Bu;
    s ^= s >> 16;
    return (float)(s >> 8) * (1.0f / 16777216.0f) + 1.0e-7f;  // (0,1)
}

// ---------------------------------------------------------------------------
// Prep: row-normalize codebook -> fp32 (gather) + f16 (WMMA); zero reg slot
// ---------------------------------------------------------------------------
__global__ void vmf_prep(const float* __restrict__ emb,
                         float* __restrict__ emb_norm,
                         _Float16* __restrict__ emb_f16,
                         float* __restrict__ reg_slot) {
    int r = blockIdx.x * blockDim.x + threadIdx.x;
    if (r == 0) *reg_slot = 0.0f;
    if (r < KCB) {
        const float* row = emb + r * DCH;
        float ss = 0.0f;
#pragma unroll
        for (int d = 0; d < DCH; ++d) { float v = row[d]; ss += v * v; }
        float inv = rsqrtf(ss);
#pragma unroll
        for (int d = 0; d < DCH; ++d) {
            float v = row[d] * inv;
            emb_norm[r * DCH + d] = v;
            emb_f16[r * DCH + d]  = (_Float16)v;
        }
    }
}

// ---------------------------------------------------------------------------
// Main: WMMA sims + streaming weighted-reservoir categorical sample
//       + fused gather + reg accumulation.
//   block = 256 threads = 8 waves; wave handles 16 tokens vs all 512 codes
// ---------------------------------------------------------------------------
__global__ __launch_bounds__(256, 1) void vmf_quant(
    const float* __restrict__ z_e,
    const float* __restrict__ emb_norm,
    const _Float16* __restrict__ emb_f16,
    const float* __restrict__ kappa_p,
    float* __restrict__ out)
{
    __shared__ _Float16 s_emb[CHUNK_CODES * DCH];   // 32 KB codebook chunk
    __shared__ int      s_idx[8][16];               // chosen code per token

    const float kappa = kappa_p[0];
    const int lane = threadIdx.x & 31;
    const int wave = threadIdx.x >> 5;
    const int l16  = lane & 15;
    const int hf   = lane >> 4;                     // 16-lane half select
    const int tile_m = (blockIdx.x * 8 + wave) * 16;

    // ---- load A fragments (16 tokens x 64 channels, f32 -> f16) ----
    // 16-bit A 16x32 layout: lanes<16 VGPR j: K = 2j+8*hf (j<4) / 16+2(j-4)+8*hf
    const int tok = tile_m + l16;
    const int bb  = tok >> 12;          // token / 4096
    const int pp  = tok & 4095;
    const float* zbase = z_e + (size_t)bb * (DCH * HWSZ) + pp;

    v16h a0, a1;
#pragma unroll
    for (int j = 0; j < 8; ++j) {
        int k = (j < 4) ? (8 * hf + 2 * j) : (16 + 8 * hf + 2 * (j - 4));
        a0[2 * j]     = (_Float16)zbase[(size_t)(k +  0) * HWSZ];
        a0[2 * j + 1] = (_Float16)zbase[(size_t)(k +  1) * HWSZ];
        a1[2 * j]     = (_Float16)zbase[(size_t)(k + 32) * HWSZ];
        a1[2 * j + 1] = (_Float16)zbase[(size_t)(k + 33) * HWSZ];
    }

    // streaming reservoir state per (row r, code-column l16) stream
    float Wsum[8], sel_sim[8];
    int   sel_idx[8];
#pragma unroll
    for (int r = 0; r < 8; ++r) { Wsum[r] = 0.0f; sel_sim[r] = 0.0f; sel_idx[r] = 0; }

    // ---- sweep codebook: 2 chunks x 16 tiles of 16 codes ----
    for (int chunk = 0; chunk < KCB / CHUNK_CODES; ++chunk) {
        __syncthreads();
        {   // stage 32KB chunk into LDS with 128-bit copies
            const uint4* src = (const uint4*)(emb_f16 + (size_t)chunk * CHUNK_CODES * DCH);
            uint4* dst = (uint4*)s_emb;
#pragma unroll
            for (int i = threadIdx.x; i < CHUNK_CODES * DCH / 8; i += 256) dst[i] = src[i];
        }
        __syncthreads();

        if (chunk == 0) {
            // warm L2/WGP$ for the next chunk while we compute this one
            // (global_prefetch_b8: one cacheline per thread covers 32KB)
            const char* nxt = (const char*)(emb_f16 + (size_t)CHUNK_CODES * DCH);
            __builtin_prefetch(nxt + threadIdx.x * 128, 0, 3);
        }

#pragma unroll 2
        for (int t = 0; t < CHUNK_CODES / 16; ++t) {
            // B fragment: column = l16, K in same table order as A
            const _Float16* erow = s_emb + (t * 16 + l16) * DCH;
            v16h b0, b1;
#pragma unroll
            for (int j = 0; j < 8; ++j) {
                int k = (j < 4) ? (8 * hf + 2 * j) : (16 + 8 * hf + 2 * (j - 4));
                b0[2 * j]     = erow[k];
                b0[2 * j + 1] = erow[k + 1];
                b1[2 * j]     = erow[k + 32];
                b1[2 * j + 1] = erow[k + 33];
            }
            v8f c = {};
            c = __builtin_amdgcn_wmma_f32_16x16x32_f16(false, a0, false, b0,
                                                       (short)0, c, false, false);
            c = __builtin_amdgcn_wmma_f32_16x16x32_f16(false, a1, false, b1,
                                                       (short)0, c, false, false);

            const int ccol = chunk * CHUNK_CODES + t * 16 + l16;
#pragma unroll
            for (int r = 0; r < 8; ++r) {
                // C layout: VGPR r -> row M = r + 8*hf, lane%16 -> column
                const int token = tile_m + r + 8 * hf;
                float sim = c[r];
                float w   = __expf(kappa * sim);       // softmax weight (1 trans)
                float W   = Wsum[r] + w;
                // A-Chao weighted reservoir: accept new item with prob w/W
                float u   = u01_hash((unsigned)(token * KCB + ccol) * 0x9E3779B9u);
                bool take = (u * W) < w;               // first item always taken
                sel_idx[r] = take ? ccol : sel_idx[r];
                sel_sim[r] = take ? sim  : sel_sim[r];
                Wsum[r] = W;
            }
        }
    }

    // ---- merge the 16 per-lane reservoirs (xor masks 1..8 stay in-half) ----
    // pairwise merge keeps exact categorical distribution: pick the 'hi'
    // element with prob W_hi / (W_lo + W_hi); seed symmetric in the pair so
    // both lanes compute identical results.
#pragma unroll
    for (int r = 0; r < 8; ++r) {
        float W  = Wsum[r];
        float sm = sel_sim[r];
        int   ix = sel_idx[r];
        const int token = tile_m + r + 8 * hf;
#pragma unroll
        for (int m = 1; m <= 8; m <<= 1) {
            float oW  = __shfl_xor(W,  m, 32);
            float osm = __shfl_xor(sm, m, 32);
            int   oix = __shfl_xor(ix, m, 32);
            bool iAmLo = (lane & m) == 0;
            float W_lo  = iAmLo ? W  : oW,  W_hi  = iAmLo ? oW  : W;
            float sm_lo = iAmLo ? sm : osm, sm_hi = iAmLo ? osm : sm;
            int   ix_lo = iAmLo ? ix : oix, ix_hi = iAmLo ? oix : ix;
            float Wt = W_lo + W_hi;
            float u  = u01_hash((unsigned)token * 747796405u
                                + (unsigned)m * 0x632BE59Bu + 0x9E3779B9u);
            bool takeHi = (u * Wt) < W_hi;
            W  = Wt;
            sm = takeHi ? sm_hi : sm_lo;
            ix = takeHi ? ix_hi : ix_lo;
        }
        sel_sim[r] = sm;
        sel_idx[r] = ix;
    }

    // ---- emit indices + reg partial (lanes 0 and 16 own rows r / r+8) ----
    if (l16 == 0) {
        float part = 0.0f;
#pragma unroll
        for (int r = 0; r < 8; ++r) {
            int token = tile_m + r + 8 * hf;
            out[IDX_OFF + token] = (float)sel_idx[r];
            s_idx[wave][r + 8 * hf] = sel_idx[r];
            part += kappa * (1.0f - sel_sim[r]);
        }
        atomicAdd(out + REG_OFF, part * (1.0f / (float)NTOK));
    }
    __syncthreads();

    // ---- gather z_q = emb_norm[idx] : lane l16 -> token, lane half -> d range
    {
        const int gtok = tile_m + l16;
        const int idx  = s_idx[wave][l16];
        const int gb   = gtok >> 12;
        const int gp   = gtok & 4095;
        float* obase = out + (size_t)gb * (DCH * HWSZ) + gp;
        const float* erow = emb_norm + idx * DCH;
#pragma unroll
        for (int j = 0; j < 32; ++j) {
            int d = hf * 32 + j;
            obase[(size_t)d * HWSZ] = erow[d];    // 64B coalesced across 16 lanes
        }
    }
}

// ---------------------------------------------------------------------------
extern "C" void kernel_launch(void* const* d_in, const int* in_sizes, int n_in,
                              void* d_out, int out_size, void* d_ws, size_t ws_size,
                              hipStream_t stream) {
    const float* z_e   = (const float*)d_in[0];
    const float* emb   = (const float*)d_in[1];
    const float* kappa = (const float*)d_in[2];
    float* out = (float*)d_out;

    float*    emb_norm = (float*)d_ws;                                   // 128 KB
    _Float16* emb_f16  = (_Float16*)((char*)d_ws + KCB * DCH * sizeof(float)); // 64 KB

    vmf_prep<<<2, 256, 0, stream>>>(emb, emb_norm, emb_f16, out + REG_OFF);
    vmf_quant<<<NTOK / 128, 256, 0, stream>>>(z_e, emb_norm, emb_f16, kappa, out);
}
